// GroupedQueryAttention_32590211842060
// MI455X (gfx1250) — compile-verified, matrix operand order checked by reading, not run on hardware
//
#include <hip/hip_runtime.h>
#include <hip/hip_bf16.h>
#include <stdint.h>

// ---------- types ----------
typedef __bf16 bf16_t;
typedef __attribute__((ext_vector_type(16))) __bf16 v16bf;
typedef __attribute__((ext_vector_type(8)))  float  v8f;
typedef __attribute__((ext_vector_type(4)))  unsigned int u32x4;

union FragU { v16bf v; u32x4 q[2]; bf16_t e[16]; };
union VF    { v8f v; float f[8]; };

#define LOG2E 1.4426950408889634f

// Contiguous 16-element bf16 fragment load (A-layout row / B-layout column
// from a pre-transposed matrix). For float sources, convert on the fly.
template <typename T>
__device__ inline v16bf load_frag(const T* p) {
  FragU f;
  if constexpr (sizeof(T) == 2) {
    f.q[0] = ((const u32x4*)p)[0];
    f.q[1] = ((const u32x4*)p)[1];
  } else {
#pragma unroll
    for (int i = 0; i < 16; ++i) f.e[i] = (bf16_t)p[i];
  }
  return f.v;
}

__device__ inline v8f wmma_bf16(v16bf a, v16bf b, v8f c) {
  return __builtin_amdgcn_wmma_f32_16x16x32_bf16(false, a, false, b,
                                                 (short)0, c, false, false);
}

// ---------- 16-lane max reduction (VALU DPP, no LDS traffic) ----------
#if __has_builtin(__builtin_amdgcn_update_dpp)
template <int CTRL>
__device__ inline float dpp_max_combine(float x) {
  int t = __builtin_amdgcn_update_dpp(0, __float_as_int(x), CTRL, 0xF, 0xF,
                                      true);
  return fmaxf(x, __int_as_float(t));
}

__device__ inline float rowmax16(float x) {
  x = dpp_max_combine<0xB1>(x);   // quad_perm [1,0,3,2]  (xor1)
  x = dpp_max_combine<0x4E>(x);   // quad_perm [2,3,0,1]  (xor2)
  x = dpp_max_combine<0x140>(x);  // row_mirror           (xor15)
  x = dpp_max_combine<0x141>(x);  // row_half_mirror      (xor7)
  return x;
}
#else
__device__ inline float rowmax16(float x) {
#pragma unroll
  for (int mask = 1; mask < 16; mask <<= 1)
    x = fmaxf(x, __shfl_xor(x, mask, 32));
  return x;
}
#endif

// ---------- weight transpose: W[k][n] f32 -> WT[n][k] bf16 ----------
__global__ __launch_bounds__(256) void transpose_w(const float* __restrict__ W,
                                                   bf16_t* __restrict__ WT,
                                                   int K, int N) {
  __shared__ float tile[32][33];
  const int tx = threadIdx.x, ty = threadIdx.y;  // (32, 8)
  const int n0 = blockIdx.x * 32, k0 = blockIdx.y * 32;
#pragma unroll
  for (int i = 0; i < 4; ++i)
    tile[ty + 8 * i][tx] = W[(size_t)(k0 + ty + 8 * i) * N + n0 + tx];
  __syncthreads();
#pragma unroll
  for (int i = 0; i < 4; ++i)
    WT[(size_t)(n0 + ty + 8 * i) * K + k0 + tx] = (bf16_t)tile[tx][ty + 8 * i];
}

// ---------- GEMM: C[M,N] = A[M,K] * BT[N,K]^T + bias ----------
// MODE 0: bf16 row-major out. MODE 1: bf16 per-head-transposed out
// (Vt[b][kh][d][s], S=2048). MODE 2: f32 row-major out.
template <int MODE, typename AT>
__global__ __launch_bounds__(256) void gemm_k(const AT* __restrict__ A,
                                              const bf16_t* __restrict__ BT,
                                              const float* __restrict__ bias,
                                              void* __restrict__ Cout,
                                              int M, int N, int K) {
  const int lane = threadIdx.x & 31, wid = threadIdx.x >> 5;
  const int half = lane >> 4, lr = lane & 15;
  const int wm = wid & 3, wn = wid >> 2;
  const int m0 = blockIdx.y * 128 + wm * 32;
  const int n0 = blockIdx.x * 128 + wn * 64;

  VF acc[2][4];
#pragma unroll
  for (int mt = 0; mt < 2; ++mt)
#pragma unroll
    for (int nt = 0; nt < 4; ++nt)
#pragma unroll
      for (int j = 0; j < 8; ++j) acc[mt][nt].f[j] = 0.f;

  for (int k0 = 0; k0 < K; k0 += 32) {
    v16bf a[2], b[4];
#pragma unroll
    for (int mt = 0; mt < 2; ++mt)
      a[mt] = load_frag(A + (size_t)(m0 + mt * 16 + lr) * K + k0 + 16 * half);
#pragma unroll
    for (int nt = 0; nt < 4; ++nt)
      b[nt] = load_frag(BT + (size_t)(n0 + nt * 16 + lr) * K + k0 + 16 * half);

    // distance-2 prefetch of the streaming operands (global_prefetch_b8)
    if (k0 + 64 < K) {
#pragma unroll
      for (int mt = 0; mt < 2; ++mt)
        __builtin_prefetch(A + (size_t)(m0 + mt * 16 + lr) * K + k0 + 64 +
                               16 * half, 0, 1);
#pragma unroll
      for (int nt = 0; nt < 4; ++nt)
        __builtin_prefetch(BT + (size_t)(n0 + nt * 16 + lr) * K + k0 + 64 +
                               16 * half, 0, 1);
    }

#pragma unroll
    for (int mt = 0; mt < 2; ++mt)
#pragma unroll
      for (int nt = 0; nt < 4; ++nt)
        acc[mt][nt].v = wmma_bf16(a[mt], b[nt], acc[mt][nt].v);
  }

#pragma unroll
  for (int mt = 0; mt < 2; ++mt)
#pragma unroll
    for (int nt = 0; nt < 4; ++nt)
#pragma unroll
      for (int j = 0; j < 8; ++j) {
        const int row = m0 + mt * 16 + j + 8 * half;
        const int col = n0 + nt * 16 + lr;
        const float val = acc[mt][nt].f[j] + bias[col];
        if (MODE == 0) {
          ((bf16_t*)Cout)[(size_t)row * N + col] = (bf16_t)val;
        } else if (MODE == 1) {
          const int kh = col >> 6, d = col & 63;
          const int bb = row >> 11, s = row & 2047;  // S = 2048
          ((bf16_t*)Cout)[((size_t)((bb * 8 + kh) * 64 + d)) * 2048 + s] =
              (bf16_t)val;
        } else {
          ((float*)Cout)[(size_t)row * N + col] = val;
        }
      }
}

// ---------- flash attention ----------
// Qp: [B][S][32*64] bf16; Kp: [B][S][8*64] bf16; Vt: [B][8][64][S] bf16
// Out: [B][S][2048] bf16.  Grid: (B*32, S/128), block 256 (8 waves x 16 rows).
__global__ __launch_bounds__(256) void attn_k(const bf16_t* __restrict__ Qp,
                                              const bf16_t* __restrict__ Kp,
                                              const bf16_t* __restrict__ Vt,
                                              bf16_t* __restrict__ Out) {
  __shared__ __align__(64) bf16_t Pbuf[8][16][32];

  const int lane = threadIdx.x & 31, wid = threadIdx.x >> 5;
  const int half = lane >> 4, lr = lane & 15;
  const int bh = blockIdx.x;
  const int b = bh >> 5, h = bh & 31, kh = h >> 2;  // 4 q-heads per kv-head
  const int q0 = blockIdx.y * 128 + wid * 16;

  // Q fragments for this wave's 16 rows: d 0..31 and 32..63
  const bf16_t* qp = Qp + ((size_t)(b * 2048 + q0 + lr)) * 2048 + h * 64;
  const v16bf a0 = load_frag(qp + 16 * half);
  const v16bf a1 = load_frag(qp + 32 + 16 * half);

  // all-ones B fragment: accumulates row-sums of P via WMMA (every column of
  // the resulting C tile holds the row sum -> no cross-lane sum reduction)
  FragU onesf;
#pragma unroll
  for (int i = 0; i < 16; ++i) onesf.e[i] = (bf16_t)1.0f;
  const v16bf ones = onesf.v;

  VF o[4], ol;
  float mrow[8], sc[8];
#pragma unroll
  for (int dn = 0; dn < 4; ++dn)
#pragma unroll
    for (int j = 0; j < 8; ++j) o[dn].f[j] = 0.f;
#pragma unroll
  for (int j = 0; j < 8; ++j) { ol.f[j] = 0.f; mrow[j] = -1e30f; }

  for (int kb = 0; kb < 2048; kb += 32) {
    // S tile: Q(16x64) . K^T(64x32) via 2 key sub-tiles, 2 d-steps each
    VF s[2];
#pragma unroll
    for (int t = 0; t < 2; ++t) {
      const int key = kb + t * 16 + lr;
      const bf16_t* kp = Kp + ((size_t)(b * 2048 + key)) * 512 + kh * 64;
      const v16bf bk0 = load_frag(kp + 16 * half);
      const v16bf bk1 = load_frag(kp + 32 + 16 * half);
#pragma unroll
      for (int j = 0; j < 8; ++j) s[t].f[j] = 0.f;
      s[t].v = wmma_bf16(a0, bk0, s[t].v);
      s[t].v = wmma_bf16(a1, bk1, s[t].v);
    }

    // online softmax over the 32 keys of this block (max via DPP, sum via WMMA)
#pragma unroll
    for (int j = 0; j < 8; ++j) {
      const float x0 = s[0].f[j] * 0.125f;  // 1/sqrt(64)
      const float x1 = s[1].f[j] * 0.125f;
      const float tm = rowmax16(fmaxf(x0, x1));
      const float mn = fmaxf(mrow[j], tm);
      sc[j] = exp2f((mrow[j] - mn) * LOG2E);
      mrow[j] = mn;
      const float p0 = exp2f((x0 - mn) * LOG2E);
      const float p1 = exp2f((x1 - mn) * LOG2E);
      // C-layout -> LDS (row j+8*half, cols lr / 16+lr)
      Pbuf[wid][j + 8 * half][lr] = (bf16_t)p0;
      Pbuf[wid][j + 8 * half][16 + lr] = (bf16_t)p1;
    }

    asm volatile("s_wait_dscnt 0" ::: "memory");

    // A-layout read of P (row lr, keys 16*half..16*half+15)
    const v16bf pa = load_frag(&Pbuf[wid][lr][16 * half]);

    // rescale running accumulators, then O += P.V and l += rowsum(P)
#pragma unroll
    for (int j = 0; j < 8; ++j) ol.f[j] *= sc[j];
    ol.v = wmma_bf16(pa, ones, ol.v);

#pragma unroll
    for (int dn = 0; dn < 4; ++dn) {
      const v16bf bv = load_frag(
          Vt + ((size_t)((b * 8 + kh) * 64 + dn * 16 + lr)) * 2048 + kb +
          16 * half);
#pragma unroll
      for (int j = 0; j < 8; ++j) o[dn].f[j] *= sc[j];
      o[dn].v = wmma_bf16(pa, bv, o[dn].v);
    }
  }

  // normalize and write bf16 output in [B][S][2048] layout
#pragma unroll
  for (int j = 0; j < 8; ++j) {
    const float inv = 1.f / ol.f[j];
    const int srow = q0 + j + 8 * half;
    const size_t base = ((size_t)(b * 2048 + srow)) * 2048 + h * 64;
#pragma unroll
    for (int dn = 0; dn < 4; ++dn)
      Out[base + dn * 16 + lr] = (bf16_t)(o[dn].f[j] * inv);
  }
}

// ---------- launcher ----------
extern "C" void kernel_launch(void* const* d_in, const int* in_sizes, int n_in,
                              void* d_out, int out_size, void* d_ws,
                              size_t ws_size, hipStream_t stream) {
  (void)in_sizes; (void)n_in; (void)out_size; (void)ws_size;
  const float* q  = (const float*)d_in[0];
  const float* k  = (const float*)d_in[1];
  const float* v  = (const float*)d_in[2];
  const float* Wq = (const float*)d_in[3];
  const float* bq = (const float*)d_in[4];
  const float* Wk = (const float*)d_in[5];
  const float* bk = (const float*)d_in[6];
  const float* Wv = (const float*)d_in[7];
  const float* bv = (const float*)d_in[8];
  const float* Wo = (const float*)d_in[9];
  const float* bo = (const float*)d_in[10];

  char* ws = (char*)d_ws;
  size_t off = 0;
  auto alloc = [&](size_t bytes) {
    void* p = ws + off;
    off += (bytes + 255) & ~(size_t)255;
    return p;
  };
  const size_t M = 4096;  // B*S
  bf16_t* WqT   = (bf16_t*)alloc((size_t)2048 * 2048 * 2);
  bf16_t* WkT   = (bf16_t*)alloc((size_t)512 * 2048 * 2);
  bf16_t* WvT   = (bf16_t*)alloc((size_t)512 * 2048 * 2);
  bf16_t* WoT   = (bf16_t*)alloc((size_t)2048 * 2048 * 2);
  bf16_t* Qp    = (bf16_t*)alloc(M * 2048 * 2);
  bf16_t* Kp    = (bf16_t*)alloc(M * 512 * 2);
  bf16_t* Vt    = (bf16_t*)alloc((size_t)2 * 512 * 2048 * 2);
  bf16_t* AttnO = (bf16_t*)alloc(M * 2048 * 2);

  const dim3 tb(32, 8);
  transpose_w<<<dim3(64, 64), tb, 0, stream>>>(Wq, WqT, 2048, 2048);
  transpose_w<<<dim3(16, 64), tb, 0, stream>>>(Wk, WkT, 2048, 512);
  transpose_w<<<dim3(16, 64), tb, 0, stream>>>(Wv, WvT, 2048, 512);
  transpose_w<<<dim3(64, 64), tb, 0, stream>>>(Wo, WoT, 2048, 2048);

  gemm_k<0, float><<<dim3(16, 32), 256, 0, stream>>>(q, WqT, bq, Qp,
                                                     4096, 2048, 2048);
  gemm_k<0, float><<<dim3(4, 32), 256, 0, stream>>>(k, WkT, bk, Kp,
                                                    4096, 512, 2048);
  gemm_k<1, float><<<dim3(4, 32), 256, 0, stream>>>(v, WvT, bv, Vt,
                                                    4096, 512, 2048);

  attn_k<<<dim3(64, 16), 256, 0, stream>>>(Qp, Kp, Vt, AttnO);

  gemm_k<2, bf16_t><<<dim3(16, 32), 256, 0, stream>>>(AttnO, WoT, bo,
                                                      (float*)d_out,
                                                      4096, 2048, 2048);
}
